// QwenTransformerBlock_68839735820819
// MI455X (gfx1250) — matrix operand order repaired and checked by reading, not compiled, not run on hardware
//
#include <hip/hip_runtime.h>

// ---------------------------------------------------------------------------
// Qwen transformer block on gfx1250 (MI455X): bf16 WMMA + async-LDS GEMMs.
// B=2 S=2048 D=2048 H=16 KV=2 HD=128 I=5504, N_REP=8.
// ---------------------------------------------------------------------------

typedef __attribute__((ext_vector_type(16))) __bf16 v16bf;
typedef __attribute__((ext_vector_type(8)))  float  v8f;

#define Bv   2
#define Sv   2048
#define Dv   2048
#define Hv   16
#define KVv  2
#define HDv  128
#define Iv   5504
#define NTOK (Bv * Sv)             // 4096
#define NQKV (Dv + 2 * KVv * HDv)  // 2560

__device__ __forceinline__ v8f wmma_bf16(v16bf a, v16bf b, v8f c) {
  return __builtin_amdgcn_wmma_f32_16x16x32_bf16(false, a, false, b, (short)0, c,
                                                 false, false);
}

// A-fragment (16xK slice, 16-bit): lane L holds row (L&15); K chunks
// {kb..kb+7, kb+16..kb+23} with kb = (L<16 ? 0 : 8).
__device__ __forceinline__ v16bf load_a_frag(const __bf16* base, int stride,
                                             int row_base, int koff, int lane) {
  const __bf16* p = base + (size_t)(row_base + (lane & 15)) * stride + koff +
                    ((lane & 16) ? 8 : 0);
  v16bf a;
#pragma unroll
  for (int e = 0; e < 8; ++e) { a[e] = p[e]; a[8 + e] = p[16 + e]; }
  return a;
}

// B-fragment (Kx16 slice of an N-major, K-contiguous matrix): lane L holds
// col (L&15); K chunks {kb..kb+7, kb+16..kb+23} with kb = (L<16 ? 0 : 8).
__device__ __forceinline__ v16bf load_b_frag(const __bf16* base, int stride,
                                             int col_base, int koff, int lane) {
  const __bf16* p = base + (size_t)(col_base + (lane & 15)) * stride + koff +
                    ((lane & 16) ? 8 : 0);
  v16bf b;
#pragma unroll
  for (int e = 0; e < 8; ++e) { b[e] = p[e]; b[8 + e] = p[16 + e]; }
  return b;
}

// ---------------------------------------------------------------------------
// Generic NT GEMM: C[M,N] = A[M,K] @ B[N,K]^T, bf16 in, fp32 accum.
// EPI 0: +bias[n] -> bf16 out
// EPI 1: +res[m,n] (fp32) -> fp32 out
// EPI 2: silu(accB0) * accB1 -> bf16 out (fused gate/up)
// Block = 128 threads (4 waves); tile 128x64x32; wave computes 32x64 strip.
// Tiles staged with GLOBAL_LOAD_ASYNC_TO_LDS_B128 into double-buffered LDS:
// DMA for tile i+1 overlaps WMMA for tile i; drained with s_wait_asynccnt.
// ---------------------------------------------------------------------------
constexpr int BM = 128, BN = 64, BK = 32, LDSK = BK + 8;

template <int EPI>
__global__ void __launch_bounds__(128)
gemm_nt_kernel(const __bf16* __restrict__ A, const __bf16* __restrict__ B0,
               const __bf16* __restrict__ B1, const float* __restrict__ aux,
               __bf16* __restrict__ outb, float* __restrict__ outf,
               int M, int N, int K) {
  __shared__ __bf16 As[2][BM * LDSK];
  __shared__ __bf16 Bs[2][BN * LDSK];
  __shared__ __bf16 B1s[2][(EPI == 2) ? (BN * LDSK) : 16];

  const int tid = threadIdx.x;
  const int lane = tid & 31, wave = tid >> 5;
  const int n0 = blockIdx.x * BN, m0 = blockIdx.y * BM;

  // per-thread async-load coordinates: A = one 32-col row per thread (128
  // rows), B = half-row per thread (64 rows x 2). Both LDS and global sides
  // are contiguous, so one base pair + INST_OFFSET covers each row chunk.
  const int arow = tid;
  const int brow = tid >> 1;
  const int bcol = (tid & 1) * 16;

  auto issue = [&](int kk, int buf) {
    {
      const unsigned la = (unsigned)(size_t)&As[buf][arow * LDSK];
      const unsigned ga = (unsigned)(((size_t)(m0 + arow) * K + kk) * 2);
      asm volatile(
          "global_load_async_to_lds_b128 %0, %1, %2\n\t"
          "global_load_async_to_lds_b128 %0, %1, %2 offset:16\n\t"
          "global_load_async_to_lds_b128 %0, %1, %2 offset:32\n\t"
          "global_load_async_to_lds_b128 %0, %1, %2 offset:48"
          :: "v"(la), "v"(ga), "s"(A) : "memory");
    }
    {
      const unsigned lb = (unsigned)(size_t)&Bs[buf][brow * LDSK + bcol];
      const unsigned gb = (unsigned)(((size_t)(n0 + brow) * K + kk + bcol) * 2);
      asm volatile(
          "global_load_async_to_lds_b128 %0, %1, %2\n\t"
          "global_load_async_to_lds_b128 %0, %1, %2 offset:16"
          :: "v"(lb), "v"(gb), "s"(B0) : "memory");
    }
    if (EPI == 2) {
      const unsigned lb = (unsigned)(size_t)&B1s[buf][brow * LDSK + bcol];
      const unsigned gb = (unsigned)(((size_t)(n0 + brow) * K + kk + bcol) * 2);
      asm volatile(
          "global_load_async_to_lds_b128 %0, %1, %2\n\t"
          "global_load_async_to_lds_b128 %0, %1, %2 offset:16"
          :: "v"(lb), "v"(gb), "s"(B1) : "memory");
    }
  };

  v8f acc[2][4] = {};
  v8f acc2[2][4] = {};

  issue(0, 0);
  int buf = 0;
  for (int k0 = 0; k0 < K; k0 += BK) {
    __syncthreads();  // all waves done reading buf^1 from previous iteration
    const bool has_next = (k0 + BK) < K;
    if (has_next) issue(k0 + BK, buf ^ 1);
    if (has_next) {
      // leave next tile's loads (6 or 8 per thread) in flight
      if (EPI == 2) asm volatile("s_wait_asynccnt 0x8" ::: "memory");
      else          asm volatile("s_wait_asynccnt 0x6" ::: "memory");
    } else {
      asm volatile("s_wait_asynccnt 0x0" ::: "memory");
    }
    __syncthreads();  // current tile visible block-wide

    const __bf16* Ab = As[buf];
    const __bf16* Bb = Bs[buf];
    v16bf bf_[4];
#pragma unroll
    for (int t = 0; t < 4; ++t) bf_[t] = load_b_frag(Bb, LDSK, t * 16, 0, lane);
#pragma unroll
    for (int mi = 0; mi < 2; ++mi) {
      v16bf a = load_a_frag(Ab, LDSK, wave * 32 + mi * 16, 0, lane);
#pragma unroll
      for (int t = 0; t < 4; ++t) {
        acc[mi][t] = wmma_bf16(a, bf_[t], acc[mi][t]);
        if (EPI == 2) {
          v16bf b1 = load_b_frag(B1s[buf], LDSK, t * 16, 0, lane);
          acc2[mi][t] = wmma_bf16(a, b1, acc2[mi][t]);
        }
      }
    }
    buf ^= 1;
  }

  const int rr = (lane & 16) ? 8 : 0;
  const int cc = lane & 15;
#pragma unroll
  for (int mi = 0; mi < 2; ++mi) {
#pragma unroll
    for (int t = 0; t < 4; ++t) {
#pragma unroll
      for (int r = 0; r < 8; ++r) {
        const int gm = m0 + wave * 32 + mi * 16 + r + rr;
        const int gn = n0 + t * 16 + cc;
        const size_t idx = (size_t)gm * N + gn;
        const float v = acc[mi][t][r];
        if (EPI == 0) {
          outb[idx] = (__bf16)(v + aux[gn]);
        } else if (EPI == 1) {
          outf[idx] = v + aux[idx];
        } else {
          const float s = v / (1.0f + __expf(-v));  // silu
          outb[idx] = (__bf16)(s * acc2[mi][t][r]);
        }
      }
    }
  }
}

// ---------------------------------------------------------------------------
// RMSNorm: fp32 in -> bf16 out (one block per token).
// ---------------------------------------------------------------------------
__global__ void __launch_bounds__(256)
rmsnorm_kernel(const float* __restrict__ x, const float* __restrict__ w,
               __bf16* __restrict__ out) {
  __shared__ float red[8];
  const int t = blockIdx.x, tid = threadIdx.x;
  const float* xr = x + (size_t)t * Dv;
  float ss = 0.f;
  for (int i = tid; i < Dv; i += 256) { const float v = xr[i]; ss += v * v; }
#pragma unroll
  for (int m = 1; m < 32; m <<= 1) ss += __shfl_xor(ss, m, 32);
  if ((tid & 31) == 0) red[tid >> 5] = ss;
  __syncthreads();
  float tot = 0.f;
#pragma unroll
  for (int wv = 0; wv < 8; ++wv) tot += red[wv];
  const float rs = rsqrtf(tot / (float)Dv + 1e-6f);
  __bf16* orow = out + (size_t)t * Dv;
  for (int i = tid; i < Dv; i += 256) orow[i] = (__bf16)(xr[i] * rs * w[i]);
}

// fp32 -> bf16 cast (weights), grid-stride.
__global__ void cast_kernel(const float* __restrict__ in,
                            __bf16* __restrict__ out, size_t n) {
  size_t i = (size_t)blockIdx.x * blockDim.x + threadIdx.x;
  const size_t stride = (size_t)gridDim.x * blockDim.x;
  for (; i < n; i += stride) out[i] = (__bf16)in[i];
}

// ---------------------------------------------------------------------------
// RoPE + layout shuffle: qkv[tok, 2560] -> q[B,H,S,HD], k[B,KV,S,HD], v[B,KV,S,HD]
// cos/sin computed analytically: inv_freq = theta^(-(d%64)/64).
// ---------------------------------------------------------------------------
__global__ void __launch_bounds__(256)
rope_kernel(const __bf16* __restrict__ qkv, __bf16* __restrict__ qr,
            __bf16* __restrict__ kr, __bf16* __restrict__ vt) {
  const int t = blockIdx.x;
  const int b = t >> 11, s = t & (Sv - 1);
  const __bf16* row = qkv + (size_t)t * NQKV;
  const int tid = threadIdx.x;
  for (int i = tid; i < Dv; i += 256) {
    const int head = i >> 7, d = i & 127, j = d & 63;
    const float fr = (float)s * __powf(10000.f, -(float)j * (1.f / 64.f));
    float c, sn;
    __sincosf(fr, &sn, &c);
    const float x0 = (float)row[i];
    const float xr = (d < 64) ? -(float)row[i + 64] : (float)row[i - 64];
    qr[((size_t)(b * Hv + head) * Sv + s) * HDv + d] = (__bf16)(x0 * c + xr * sn);
  }
  {
    const int i = tid;  // 256 threads cover KV*HD = 256 elems
    const int head = i >> 7, d = i & 127, j = d & 63;
    const float fr = (float)s * __powf(10000.f, -(float)j * (1.f / 64.f));
    float c, sn;
    __sincosf(fr, &sn, &c);
    const float x0 = (float)row[Dv + i];
    const float xr = (d < 64) ? -(float)row[Dv + i + 64] : (float)row[Dv + i - 64];
    kr[((size_t)(b * KVv + head) * Sv + s) * HDv + d] = (__bf16)(x0 * c + xr * sn);
    vt[((size_t)(b * KVv + head) * Sv + s) * HDv + d] = row[Dv + KVv * HDv + i];
  }
}

// ---------------------------------------------------------------------------
// Flash attention (causal, GQA N_REP=8). Grid (S/64, H, B), 128 threads.
// Wave w owns Q rows [qt*64+16w, +16). K/V 32-key tiles staged in LDS;
// QK^T and P·V both via v_wmma_f32_16x16x32_bf16 with online softmax.
// ---------------------------------------------------------------------------
__global__ void __launch_bounds__(128)
attn_kernel(const __bf16* __restrict__ Q, const __bf16* __restrict__ Kc,
            const __bf16* __restrict__ Vc, __bf16* __restrict__ O) {
  constexpr int SK = HDv + 8;  // K-tile row stride (keys x hd)
  constexpr int SV = 32 + 8;   // Vt / P row stride
  __shared__ __bf16 Ks[32 * SK];
  __shared__ __bf16 Vt[HDv * SV];     // transposed: [hd][key]
  __shared__ __bf16 Ps[4 * 16 * SV];  // per-wave P staging

  const int tid = threadIdx.x, lane = tid & 31, wave = tid >> 5;
  const int b = blockIdx.z, h = blockIdx.y, qt = blockIdx.x;
  const int kvh = h >> 3;  // N_REP = 8
  const int qbase = qt * 64 + wave * 16;
  const int rr = (lane & 16) ? 8 : 0;
  const int cc = lane & 15;

  // Q fragments (16x128) held in registers, loaded directly in A-frag layout.
  v16bf qf[4];
  {
    const __bf16* qp = Q + ((size_t)(b * Hv + h) * Sv + qbase + (lane & 15)) * HDv +
                       ((lane & 16) ? 8 : 0);
#pragma unroll
    for (int f = 0; f < 4; ++f) {
      const __bf16* p = qp + f * 32;
#pragma unroll
      for (int e = 0; e < 8; ++e) { qf[f][e] = p[e]; qf[f][8 + e] = p[16 + e]; }
    }
  }

  v8f o[8] = {};
  float mrow[8], lrow[8];
#pragma unroll
  for (int r = 0; r < 8; ++r) { mrow[r] = -3.0e38f; lrow[r] = 0.f; }

  const __bf16* kbp = Kc + (size_t)(b * KVv + kvh) * Sv * HDv;
  const __bf16* vbp = Vc + (size_t)(b * KVv + kvh) * Sv * HDv;
  const int kend = qt * 64 + 64;
  const float scale = 0.08838834764831845f;  // 1/sqrt(128)

  for (int kb = 0; kb < kend; kb += 32) {
    __syncthreads();
    {  // cooperative K/V tile load (V stored transposed for B-frag reads)
      const int row = tid >> 2, colb = (tid & 3) * 32;
      const __bf16* g = kbp + (size_t)(kb + row) * HDv + colb;
#pragma unroll
      for (int j = 0; j < 4; ++j)
        *(uint4*)&Ks[row * SK + colb + 8 * j] = *(const uint4*)(g + 8 * j);
      const __bf16* gv = vbp + (size_t)(kb + row) * HDv + colb;
#pragma unroll
      for (int j4 = 0; j4 < 4; ++j4) {
        uint4 tmp = *(const uint4*)(gv + 8 * j4);
        const __bf16* tb = (const __bf16*)&tmp;
#pragma unroll
        for (int j = 0; j < 8; ++j)
          Vt[(colb + 8 * j4 + j) * SV + row] = tb[j];
      }
    }
    __syncthreads();

    if (kb <= qbase + 15) {  // wave-uniform causal block skip
      v8f sacc[2] = {};
#pragma unroll
      for (int f = 0; f < 4; ++f) {  // K-dim of QK^T = HD = 4 x 32
        v16bf bk0 = load_b_frag(Ks, SK, 0, f * 32, lane);
        sacc[0] = wmma_bf16(qf[f], bk0, sacc[0]);
        v16bf bk1 = load_b_frag(Ks, SK, 16, f * 32, lane);
        sacc[1] = wmma_bf16(qf[f], bk1, sacc[1]);
      }
      float s0[8], s1[8];
#pragma unroll
      for (int r = 0; r < 8; ++r) {
        const int grow = qbase + r + rr;
        s0[r] = sacc[0][r] * scale + ((kb + cc) > grow ? -1.0e9f : 0.f);
        s1[r] = sacc[1][r] * scale + ((kb + 16 + cc) > grow ? -1.0e9f : 0.f);
      }
      __bf16* pw = &Ps[wave * 16 * SV];
#pragma unroll
      for (int r = 0; r < 8; ++r) {
        float rm = fmaxf(s0[r], s1[r]);
#pragma unroll
        for (int m = 1; m <= 8; m <<= 1) rm = fmaxf(rm, __shfl_xor(rm, m, 32));
        const float mn = fmaxf(mrow[r], rm);
        const float corr = __expf(mrow[r] - mn);
        const float p0 = __expf(s0[r] - mn), p1 = __expf(s1[r] - mn);
        float rs = p0 + p1;
#pragma unroll
        for (int m = 1; m <= 8; m <<= 1) rs += __shfl_xor(rs, m, 32);
        lrow[r] = lrow[r] * corr + rs;
        mrow[r] = mn;
#pragma unroll
        for (int nt = 0; nt < 8; ++nt) o[nt][r] *= corr;
        pw[(r + rr) * SV + cc] = (__bf16)p0;
        pw[(r + rr) * SV + cc + 16] = (__bf16)p1;
      }
      // wave-private LDS RAW: drain DS counter before reading P back.
      asm volatile("s_wait_dscnt 0x0" ::: "memory");
      v16bf pa = load_a_frag(pw, SV, 0, 0, lane);
#pragma unroll
      for (int nt = 0; nt < 8; ++nt) {
        v16bf bv = load_b_frag(Vt, SV, nt * 16, 0, lane);
        o[nt] = wmma_bf16(pa, bv, o[nt]);
      }
    }
  }

  // epilogue: O /= l, write [B,S,H*HD] so heads merge into D directly
#pragma unroll
  for (int r = 0; r < 8; ++r) {
    const float inv = 1.f / lrow[r];
    const int grow = qbase + r + rr;
    __bf16* op = O + ((size_t)b * Sv + grow) * Dv + h * HDv + cc;
#pragma unroll
    for (int nt = 0; nt < 8; ++nt) op[nt * 16] = (__bf16)(o[nt][r] * inv);
  }
}

// ---------------------------------------------------------------------------
extern "C" void kernel_launch(void* const* d_in, const int* in_sizes, int n_in,
                              void* d_out, int out_size, void* d_ws,
                              size_t ws_size, hipStream_t stream) {
  const float* x   = (const float*)d_in[0];
  // d_in[1] = mask (unused: causal mask applied analytically)
  const float* ln1 = (const float*)d_in[2];
  const float* wq  = (const float*)d_in[3];
  const float* bq  = (const float*)d_in[4];
  const float* wk  = (const float*)d_in[5];
  const float* bk  = (const float*)d_in[6];
  const float* wv  = (const float*)d_in[7];
  const float* bv  = (const float*)d_in[8];
  const float* wo  = (const float*)d_in[9];
  const float* ln2 = (const float*)d_in[10];
  const float* wg  = (const float*)d_in[11];
  const float* wu  = (const float*)d_in[12];
  const float* wd  = (const float*)d_in[13];
  float* out = (float*)d_out;

  char* ws = (char*)d_ws;
  size_t off = 0;
  auto alloc = [&](size_t bytes) -> void* {
    off = (off + 255) & ~(size_t)255;
    void* p = ws + off;
    off += bytes;
    return p;
  };

  __bf16* wqkv_b = (__bf16*)alloc((size_t)NQKV * Dv * 2);
  float*  bqkv_f = (float*)alloc((size_t)NQKV * 4);
  __bf16* wo_b   = (__bf16*)alloc((size_t)Dv * Dv * 2);
  __bf16* wg_b   = (__bf16*)alloc((size_t)Iv * Dv * 2);
  __bf16* wu_b   = (__bf16*)alloc((size_t)Iv * Dv * 2);
  __bf16* wd_b   = (__bf16*)alloc((size_t)Dv * Iv * 2);
  __bf16* h1     = (__bf16*)alloc((size_t)NTOK * Dv * 2);
  __bf16* qkv    = (__bf16*)alloc((size_t)NTOK * NQKV * 2);
  __bf16* qr     = (__bf16*)alloc((size_t)Bv * Hv * Sv * HDv * 2);
  __bf16* kr     = (__bf16*)alloc((size_t)Bv * KVv * Sv * HDv * 2);
  __bf16* vt     = (__bf16*)alloc((size_t)Bv * KVv * Sv * HDv * 2);
  __bf16* att    = (__bf16*)alloc((size_t)NTOK * Dv * 2);
  float*  h2     = (float*)alloc((size_t)NTOK * Dv * 4);
  __bf16* g1     = (__bf16*)alloc((size_t)NTOK * Dv * 2);
  __bf16* act    = (__bf16*)alloc((size_t)NTOK * Iv * 2);

  // --- weight casts (fp32 -> bf16), QKV packed N-major: [q:2048 | k:256 | v:256]
  cast_kernel<<<2048, 256, 0, stream>>>(wq, wqkv_b, (size_t)Dv * Dv);
  cast_kernel<<<512, 256, 0, stream>>>(wk, wqkv_b + (size_t)Dv * Dv,
                                       (size_t)KVv * HDv * Dv);
  cast_kernel<<<512, 256, 0, stream>>>(wv, wqkv_b + (size_t)(Dv + KVv * HDv) * Dv,
                                       (size_t)KVv * HDv * Dv);
  hipMemcpyAsync(bqkv_f, bq, Dv * 4, hipMemcpyDeviceToDevice, stream);
  hipMemcpyAsync(bqkv_f + Dv, bk, KVv * HDv * 4, hipMemcpyDeviceToDevice, stream);
  hipMemcpyAsync(bqkv_f + Dv + KVv * HDv, bv, KVv * HDv * 4,
                 hipMemcpyDeviceToDevice, stream);
  cast_kernel<<<2048, 256, 0, stream>>>(wo, wo_b, (size_t)Dv * Dv);
  cast_kernel<<<4096, 256, 0, stream>>>(wg, wg_b, (size_t)Iv * Dv);
  cast_kernel<<<4096, 256, 0, stream>>>(wu, wu_b, (size_t)Iv * Dv);
  cast_kernel<<<4096, 256, 0, stream>>>(wd, wd_b, (size_t)Dv * Iv);

  // --- 1) h = rmsnorm(x, ln1)  (bf16)
  rmsnorm_kernel<<<NTOK, 256, 0, stream>>>(x, ln1, h1);

  // --- 2) fused QKV GEMM + bias -> bf16
  gemm_nt_kernel<0><<<dim3(NQKV / BN, NTOK / BM), 128, 0, stream>>>(
      h1, wqkv_b, nullptr, bqkv_f, qkv, nullptr, NTOK, NQKV, Dv);

  // --- 3) RoPE + head-major layouts
  rope_kernel<<<NTOK, 256, 0, stream>>>(qkv, qr, kr, vt);

  // --- 4) causal GQA flash attention
  attn_kernel<<<dim3(Sv / 64, Hv, Bv), 128, 0, stream>>>(qr, kr, vt, att);

  // --- 5) h2 = x + attn @ wo^T  (fp32)
  gemm_nt_kernel<1><<<dim3(Dv / BN, NTOK / BM), 128, 0, stream>>>(
      att, wo_b, nullptr, x, nullptr, h2, NTOK, Dv, Dv);

  // --- 6) g = rmsnorm(h2, ln2)  (bf16)
  rmsnorm_kernel<<<NTOK, 256, 0, stream>>>(h2, ln2, g1);

  // --- 7) act = silu(g @ wg^T) * (g @ wu^T)  (fused dual-GEMM, bf16)
  gemm_nt_kernel<2><<<dim3(Iv / BN, NTOK / BM), 128, 0, stream>>>(
      g1, wg_b, wu_b, nullptr, act, nullptr, NTOK, Iv, Dv);

  // --- 8) out = h2 + act @ wd^T  (fp32)
  gemm_nt_kernel<1><<<dim3(Dv / BN, NTOK / BM), 128, 0, stream>>>(
      act, wd_b, nullptr, h2, nullptr, out, NTOK, Dv, Iv);
}
